// slice_model_80075370267325
// MI455X (gfx1250) — matrix operand order repaired and checked by reading, not compiled, not run on hardware
//
#include <hip/hip_runtime.h>
#include <stdint.h>

// ---------------- problem constants (from reference) ----------------
#define NUM_NODES 65536   // B*N
#define FEAT_C    128
#define NGRAPH    16
#define NLEN      4096
#define LOUT      4094    // NLEN - 2 (VALID conv, k=3)

// ---------------- CDNA5 WMMA types ----------------
typedef __bf16 bf16_t;
typedef bf16_t v16bf __attribute__((ext_vector_type(16)));
typedef float  v8f   __attribute__((ext_vector_type(8)));

static __device__ __forceinline__ v8f wmma_bf16(v16bf a, v16bf b, v8f c) {
  // D = A(16x32 bf16) x B(32x16 bf16) + C(16x16 f32)
  return __builtin_amdgcn_wmma_f32_16x16x32_bf16(false, a, false, b, (short)0, c, false, false);
}

// f32 -> bf16 (round to nearest even)
static __device__ __forceinline__ unsigned short f2bf(float f) {
  unsigned u = __float_as_uint(f);
  u += 0x7FFFu + ((u >> 16) & 1u);
  return (unsigned short)(u >> 16);
}

// load 16 bf16 (as raw ushorts) from two 16B-aligned addresses into a v16bf
static __device__ __forceinline__ v16bf ldbf16x16(const unsigned short* p0,
                                                  const unsigned short* p1) {
  union { uint4 q[2]; v16bf v; } u;
  u.q[0] = *(const uint4*)p0;
  u.q[1] = *(const uint4*)p1;
  return u.v;
}

// ---------------- CDNA5 async global->LDS copy (ASYNCcnt path) -------------
// One 16B chunk per lane: LDS byte offset (within dynamic group segment,
// which starts at LDS offset 0) in a VGPR, 64-bit global address in a pair.
static __device__ __forceinline__ void async_copy16(unsigned lds_off,
                                                    const void* gaddr) {
  asm volatile("global_load_async_to_lds_b128 %0, %1, off"
               :: "v"(lds_off), "v"((unsigned long long)(uintptr_t)gaddr)
               : "memory");
}
static __device__ __forceinline__ void async_wait0() {
  asm volatile("s_wait_asynccnt 0x0" ::: "memory");
}

// ---- order-preserving float<->uint encoding for native atomic max ----
static __device__ __forceinline__ unsigned enc_key(float f) {
  unsigned b = __float_as_uint(f);
  return (b & 0x80000000u) ? ~b : (b | 0x80000000u);
}
static __device__ __forceinline__ float dec_key(unsigned k) {
  if (k == 0u) return 0.0f;                       // empty segment -> 0 (PyG)
  unsigned b = (k & 0x80000000u) ? (k & 0x7FFFFFFFu) : ~k;
  if ((b & 0x7F800000u) == 0x7F800000u) return 0.0f;  // non-finite -> 0
  return __uint_as_float(b);
}

// =====================================================================
// 1) scatter-max: agg[dst, c] = max over edges of feature[src, c]
//    one wave per edge, 4 channels per lane, global_atomic_max_u32
// =====================================================================
__global__ void __launch_bounds__(256) scatter_max_k(
    const float* __restrict__ feat, const int* __restrict__ src,
    const int* __restrict__ dst, unsigned* __restrict__ agg, int E)
{
  const int wv   = (blockIdx.x * blockDim.x + threadIdx.x) >> 5;
  const int lane = threadIdx.x & 31;
  const int nwv  = (gridDim.x * blockDim.x) >> 5;
  for (int e = wv; e < E; e += nwv) {
    const int s = src[e], d = dst[e];
    const float4 f4 = *(const float4*)(feat + (size_t)s * FEAT_C + lane * 4);
    unsigned* a = agg + (size_t)d * FEAT_C + lane * 4;
    atomicMax(a + 0, enc_key(f4.x));
    atomicMax(a + 1, enc_key(f4.y));
    atomicMax(a + 2, enc_key(f4.z));
    atomicMax(a + 3, enc_key(f4.w));
  }
}

// =====================================================================
// 2) build Xin[node][0:128]=decode(agg) bf16, Xin[node][128:256]=feature
//    (node-major, stride 256 -> K-contiguous for WMMA B/A loads)
// =====================================================================
__global__ void __launch_bounds__(256) build_xin_k(
    const unsigned* __restrict__ agg, const float* __restrict__ feat,
    unsigned short* __restrict__ X)
{
  const size_t i = (size_t)blockIdx.x * blockDim.x + threadIdx.x; // over N*128
  const size_t n = i >> 7;
  const int    c = (int)(i & 127);
  X[n * 256 + c]       = f2bf(dec_key(agg[i]));
  X[n * 256 + 128 + c] = f2bf(feat[i]);
}

// =====================================================================
// 3) weight prep (all bf16):
//    WcT[co][k] : k<128 -> lin_l_w[co][k], else lin_r_w[co][k-128]  (K-contig)
//    Wt1[t][co][ci] <- conv1_w[co][ci][t]  (K=ci contiguous)
//    Wt2[t][co][ci] <- conv2_w[co][ci][t]
// =====================================================================
__global__ void __launch_bounds__(256) prep_w_k(
    const float* __restrict__ ll, const float* __restrict__ lr,
    const float* __restrict__ c1, const float* __restrict__ c2,
    unsigned short* __restrict__ WcT, unsigned short* __restrict__ Wt1,
    unsigned short* __restrict__ Wt2)
{
  int i = blockIdx.x * blockDim.x + threadIdx.x;
  if (i < 32768) {                              // WcT: 128 x 256
    const int co = i >> 8, k = i & 255;
    WcT[i] = f2bf(k < 128 ? ll[co * 128 + k] : lr[co * 128 + (k - 128)]);
  } else if (i < 32768 + 49152) {               // Wt1: 3 x 128 x 128
    const int j = i - 32768;
    const int t = j / 16384, r = j % 16384;
    const int co = r >> 7, ci = r & 127;
    Wt1[j] = f2bf(c1[(co * 128 + ci) * 3 + t]);
  } else if (i < 32768 + 49152 + 196608) {      // Wt2: 3 x 256 x 256
    const int j = i - 81920;
    const int t = j / 65536, r = j % 65536;
    const int co = r >> 8, ci = r & 255;
    Wt2[j] = f2bf(c2[(co * 256 + ci) * 3 + t]);
  }
}

// =====================================================================
// 4) SAGE GEMM + L2 row-normalize.
//    sage[n, co] = sum_{k<256} Xin[n][k] * WcT[co][k] + bias[co]
//    WcT (64 KB) is async-staged into LDS once per block; B tiles then
//    come from ds_load_b128 instead of global (removes the
//    load->s_wait_loadcnt->wmma serialization seen in round 0 disasm).
//    Each wave: 16 nodes x 128 co (8 WMMA n-tiles), K loop 8 x 32.
//    Normalized bf16 result overwrites Xin[n][0:128] (conv branch input).
// =====================================================================
__global__ void __launch_bounds__(256) sage_k(
    unsigned short* __restrict__ X, const unsigned short* __restrict__ WcT,
    const float* __restrict__ bias)
{
  extern __shared__ __align__(16) unsigned char dynLds[];
  unsigned short* sW = (unsigned short*)dynLds;   // 128*256 bf16 @ LDS offset 0

  const int tid = threadIdx.x;
  // async-stage 64 KiB of weights: 4096 x 16B chunks, ASYNCcnt-tracked
  for (int i = tid; i < 4096; i += 256)
    async_copy16((unsigned)(i * 16), (const unsigned char*)WcT + (size_t)i * 16);
  async_wait0();
  __syncthreads();

  const int wv = tid >> 5, lane = tid & 31, half = lane >> 4, ln = lane & 15;
  const size_t nodeBase = (size_t)blockIdx.x * 128 + (size_t)wv * 16;

  v8f acc[8];
  #pragma unroll
  for (int nt = 0; nt < 8; ++nt)
    #pragma unroll
    for (int v = 0; v < 8; ++v) acc[nt][v] = 0.0f;

  const unsigned short* arow = X + (nodeBase + ln) * 256;
  for (int kb = 0; kb < 8; ++kb) {
    // A: 16 nodes x 32 k (per-lane row = ln, K contiguous at half*16)
    const v16bf A = ldbf16x16(arow + kb * 32 + half * 16,
                              arow + kb * 32 + half * 16 + 8);
    #pragma unroll
    for (int nt = 0; nt < 8; ++nt) {
      // B: 32 k x 16 co (per-lane col = ln, K contiguous at half*16), from LDS
      const unsigned short* brow =
          sW + (size_t)(nt * 16 + ln) * 256 + kb * 32 + half * 16;
      const v16bf Bm = ldbf16x16(brow, brow + 8);
      acc[nt] = wmma_bf16(A, Bm, acc[nt]);
    }
  }

  // bias + per-row sum of squares (row r = v + 8*half, cols across lanes/nt)
  float ss[8];
  #pragma unroll
  for (int v = 0; v < 8; ++v) ss[v] = 0.0f;
  #pragma unroll
  for (int nt = 0; nt < 8; ++nt) {
    const float bv = bias[nt * 16 + ln];
    #pragma unroll
    for (int v = 0; v < 8; ++v) {
      const float y = acc[nt][v] + bv;
      acc[nt][v] = y;
      ss[v] += y * y;
    }
  }
  #pragma unroll
  for (int m = 1; m < 16; m <<= 1)
    #pragma unroll
    for (int v = 0; v < 8; ++v) ss[v] += __shfl_xor(ss[v], m, 32);

  float scl[8];
  #pragma unroll
  for (int v = 0; v < 8; ++v) scl[v] = 1.0f / fmaxf(sqrtf(ss[v]), 1e-12f);

  #pragma unroll
  for (int nt = 0; nt < 8; ++nt)
    #pragma unroll
    for (int v = 0; v < 8; ++v)
      X[(nodeBase + v + 8 * half) * 256 + nt * 16 + ln] =
          f2bf(acc[nt][v] * scl[v]);
}

// =====================================================================
// 5) Conv1d(k=3) as WMMA GEMM, shared for both branches & both passes.
//    y[b][co][l] = sum_t sum_ci W[t][co][ci] * x[b][ci][l+t] + bias[co]
//    Weight tile (3 taps x 16 co x Cin) async-staged into LDS.
//    pass 0: accumulate per-channel sum / sumsq (BN training stats)
//    pass 1: y = scale*y + shift; relu; maxpool over l -> pool[b][co]
//    Block: (co-tile of 16, graph b); 8 waves stride over 256 l-tiles.
// =====================================================================
__global__ void __launch_bounds__(256) conv_bn_k(
    const unsigned short* __restrict__ X, const unsigned short* __restrict__ W,
    const float* __restrict__ bias, int Cin, int Co, int pass,
    float* __restrict__ gSum, float* __restrict__ gSq,
    const float* __restrict__ scale, const float* __restrict__ shift,
    float* __restrict__ pool)
{
  extern __shared__ __align__(16) unsigned char dynLds[];
  unsigned short* lW = (unsigned short*)dynLds;      // 3*16*Cin bf16 @ offset 0
  float* sA = (float*)(dynLds + 96 * Cin);           // 96*Cin = 3*16*Cin*2 B
  float* sB = sA + 16;

  const int tid    = threadIdx.x;
  const int coBase = blockIdx.x * 16;
  const int b      = blockIdx.y;

  if (tid < 16) { sA[tid] = 0.0f; sB[tid] = 0.0f; }

  // async-stage weight tile: per tap, 16 co rows x Cin (K-contiguous)
  const int chunksPerTap = (32 * Cin) >> 4;  // bytes per tap / 16
  for (int i = tid; i < 3 * chunksPerTap; i += 256) {
    const int t = i / chunksPerTap;
    const int r = i - t * chunksPerTap;
    async_copy16((unsigned)(t * 32 * Cin + r * 16),
                 (const unsigned char*)(W + (size_t)(t * Co + coBase) * Cin) +
                     (size_t)r * 16);
  }
  async_wait0();
  __syncthreads();

  const int wv = tid >> 5, lane = tid & 31, half = lane >> 4, ln = lane & 15;
  const int kBlocks = Cin >> 5;

  float r0[8], r1[8];
  #pragma unroll
  for (int v = 0; v < 8; ++v) { r0[v] = 0.0f; r1[v] = 0.0f; }

  for (int lt = wv; lt < 256; lt += 8) {
    const int lBase = lt << 4;
    v8f acc;
    #pragma unroll
    for (int v = 0; v < 8; ++v) acc[v] = 0.0f;

    #pragma unroll
    for (int t = 0; t < 3; ++t) {
      int node = b * NLEN + lBase + ln + t;          // B column l = lBase+ln
      if (node > NUM_NODES - 1) node = NUM_NODES - 1; // clamp (masked later)
      const unsigned short* xrow = X + (size_t)node * 256;
      const unsigned short* wrow = lW + (t * 16 + ln) * Cin;  // A row co=ln
      for (int kb = 0; kb < kBlocks; ++kb) {
        const v16bf A = ldbf16x16(wrow + kb * 32 + half * 16,
                                  wrow + kb * 32 + half * 16 + 8);
        const v16bf Bm = ldbf16x16(xrow + kb * 32 + half * 16,
                                   xrow + kb * 32 + half * 16 + 8);
        acc = wmma_bf16(A, Bm, acc);
      }
    }

    const bool valid = (lBase + ln) < LOUT;
    #pragma unroll
    for (int v = 0; v < 8; ++v) {
      const int co = coBase + v + 8 * half;
      const float y = acc[v] + bias[co];
      if (pass == 0) {
        const float yv = valid ? y : 0.0f;
        r0[v] += yv;
        r1[v] += yv * yv;
      } else {
        float z = fmaf(scale[co], y, shift[co]);
        z = fmaxf(z, 0.0f);
        if (!valid) z = 0.0f;   // relu output >= 0, so 0 is neutral for max
        r0[v] = fmaxf(r0[v], z);
      }
    }
  }

  // reduce over the 16 lanes of each half (matches D-matrix column layout)
  #pragma unroll
  for (int m = 1; m < 16; m <<= 1) {
    #pragma unroll
    for (int v = 0; v < 8; ++v) {
      const float a = __shfl_xor(r0[v], m, 32);
      if (pass == 0) { r0[v] += a; r1[v] += __shfl_xor(r1[v], m, 32); }
      else           { r0[v] = fmaxf(r0[v], a); }
    }
  }
  if (ln == 0) {
    #pragma unroll
    for (int v = 0; v < 8; ++v) {
      const int idx = v + 8 * half;
      if (pass == 0) { atomicAdd(&sA[idx], r0[v]); atomicAdd(&sB[idx], r1[v]); }
      else           { atomicMax((int*)&sA[idx], __float_as_int(r0[v])); }
    }
  }
  __syncthreads();
  if (tid < 16) {
    if (pass == 0) {
      atomicAdd(&gSum[coBase + tid], sA[tid]);
      atomicAdd(&gSq[coBase + tid], sB[tid]);
    } else {
      pool[(size_t)b * Co + coBase + tid] = sA[tid];
    }
  }
}

// =====================================================================
// 6) BN finalize: scale = g*rsqrt(var+eps), shift = b - mean*scale
//    stats layout (floats): [s1 128][q1 128][sc1 128][sh1 128]
//                           [s2 256][q2 256][sc2 256][sh2 256]
// =====================================================================
__global__ void bn_fin_k(float* __restrict__ st,
                         const float* __restrict__ g1, const float* __restrict__ b1,
                         const float* __restrict__ g2, const float* __restrict__ b2)
{
  const float inv = 1.0f / ((float)NGRAPH * (float)LOUT);
  const int i = threadIdx.x;
  if (i < 128) {
    const float m = st[i] * inv;
    const float v = st[128 + i] * inv - m * m;
    const float a = g1[i] * rsqrtf(v + 1e-5f);
    st[256 + i] = a;
    st[384 + i] = b1[i] - m * a;
  } else if (i < 384) {
    const int c = i - 128;
    const float m = st[512 + c] * inv;
    const float v = st[768 + c] * inv - m * m;
    const float a = g2[c] * rsqrtf(v + 1e-5f);
    st[1024 + c] = a;
    st[1280 + c] = b2[c] - m * a;
  }
}

// =====================================================================
// 7) heads: out[b,j] = (pool1[b]·lin1_w[j] + lin1_b[j])
//                    * (pool2[b]·lin2_w[j] + lin2_b[j])
// =====================================================================
__global__ void heads_k(const float* __restrict__ p1, const float* __restrict__ p2,
                        const float* __restrict__ w1, const float* __restrict__ b1,
                        const float* __restrict__ w2, const float* __restrict__ b2,
                        float* __restrict__ out)
{
  const int i = threadIdx.x;
  if (i >= 32) return;
  const int b = i >> 1, j = i & 1;
  float a = b1[j];
  for (int c = 0; c < 128; ++c) a += p1[b * 128 + c] * w1[j * 128 + c];
  float d = b2[j];
  for (int c = 0; c < 256; ++c) d += p2[b * 256 + c] * w2[j * 256 + c];
  out[i] = a * d;
}

// =====================================================================
extern "C" void kernel_launch(void* const* d_in, const int* in_sizes, int n_in,
                              void* d_out, int out_size, void* d_ws, size_t ws_size,
                              hipStream_t stream)
{
  (void)n_in; (void)out_size; (void)ws_size;
  const float* feature = (const float*)d_in[0];
  const int*   topo    = (const int*)d_in[1];
  /* d_in[2] = batch (unused: graph id is node/4096 by construction) */
  const float* lin_l_w = (const float*)d_in[3];
  const float* lin_l_b = (const float*)d_in[4];
  const float* lin_r_w = (const float*)d_in[5];
  const float* conv1_w = (const float*)d_in[6];
  const float* conv1_b = (const float*)d_in[7];
  const float* bn1_g   = (const float*)d_in[8];
  const float* bn1_b   = (const float*)d_in[9];
  const float* conv2_w = (const float*)d_in[10];
  const float* conv2_b = (const float*)d_in[11];
  const float* bn2_g   = (const float*)d_in[12];
  const float* bn2_b   = (const float*)d_in[13];
  const float* lin1_w  = (const float*)d_in[14];
  const float* lin1_b  = (const float*)d_in[15];
  const float* lin2_w  = (const float*)d_in[16];
  const float* lin2_b  = (const float*)d_in[17];
  const int E = in_sizes[1] / 2;

  // workspace layout (~64.6 MiB total)
  char* ws = (char*)d_ws;
  unsigned*       agg   = (unsigned*)(ws + 0);                 // 32 MiB
  unsigned short* X     = (unsigned short*)(ws + 33554432);    // 32 MiB
  unsigned short* WcT   = (unsigned short*)(ws + 67108864);    // 64 KiB
  unsigned short* Wt1   = (unsigned short*)(ws + 67174400);    // 96 KiB
  unsigned short* Wt2   = (unsigned short*)(ws + 67272704);    // 384 KiB
  float*          st    = (float*)(ws + 67665920);             // 6 KiB
  float*          pool1 = (float*)(ws + 67672064);             // 8 KiB
  float*          pool2 = (float*)(ws + 67680256);             // 16 KiB

  hipMemsetAsync(agg, 0, (size_t)NUM_NODES * FEAT_C * 4, stream);
  hipMemsetAsync(st, 0, 1536 * 4, stream);

  scatter_max_k<<<4096, 256, 0, stream>>>(feature, topo, topo + E, agg, E);
  build_xin_k<<<(NUM_NODES * FEAT_C) / 256, 256, 0, stream>>>(agg, feature, X);
  prep_w_k<<<1088, 256, 0, stream>>>(lin_l_w, lin_r_w, conv1_w, conv2_w,
                                     WcT, Wt1, Wt2);
  sage_k<<<512, 256, 65536, stream>>>(X, WcT, lin_l_b);

  // BN stats pass (conv recompute strategy: cheaper than 100 MB round-trip)
  conv_bn_k<<<dim3(8, 16), 256, 96 * 128 + 128, stream>>>(
      X, Wt1, conv1_b, 128, 128, 0, st + 0, st + 128,
      nullptr, nullptr, nullptr);
  conv_bn_k<<<dim3(16, 16), 256, 96 * 256 + 128, stream>>>(
      X, Wt2, conv2_b, 256, 256, 0, st + 512, st + 768,
      nullptr, nullptr, nullptr);
  bn_fin_k<<<1, 384, 0, stream>>>(st, bn1_g, bn1_b, bn2_g, bn2_b);
  // BN + relu + maxpool pass
  conv_bn_k<<<dim3(8, 16), 256, 96 * 128 + 128, stream>>>(
      X, Wt1, conv1_b, 128, 128, 1, nullptr, nullptr,
      st + 256, st + 384, pool1);
  conv_bn_k<<<dim3(16, 16), 256, 96 * 256 + 128, stream>>>(
      X, Wt2, conv2_b, 256, 256, 1, nullptr, nullptr,
      st + 1024, st + 1280, pool2);

  heads_k<<<1, 64, 0, stream>>>(pool1, pool2, lin1_w, lin1_b, lin2_w, lin2_b,
                                (float*)d_out);
}
